// FusionBlock_21784074125888
// MI455X (gfx1250) — matrix operand order repaired and checked by reading, not compiled, not run on hardware
//
#include <hip/hip_runtime.h>

typedef __attribute__((ext_vector_type(16))) _Float16 v16h;
typedef __attribute__((ext_vector_type(4)))  _Float16 h4;
typedef __attribute__((ext_vector_type(8)))  float    v8f;

#define B_ 32
#define T_ 10
#define D_ 10
#define U_ 50
#define P_ 50
#define E_ 256

// ---------- wave32 reduction ----------
__device__ __forceinline__ float wave_sum(float v) {
#pragma unroll
  for (int off = 16; off > 0; off >>= 1) v += __shfl_xor(v, off, 32);
  return v;
}

// ---------- WMMA fragment loaders (wave32, 16x16x32 f16) ----------
// A fragment: 16x32 (MxK), source row-major with leading dim ld (elements).
// lanes 0-15:  row=lane,    elems j = K j
// lanes 16-31: row=lane-16, elems j = K 16+j
__device__ __forceinline__ v16h load_a_lds(const _Float16* p0, int ld) {
  const int lane = threadIdx.x & 31;
  const _Float16* p = p0 + (lane & 15) * ld + ((lane < 16) ? 0 : 16);
  v16h a;
#pragma unroll
  for (int j = 0; j < 16; ++j) a[j] = p[j];
  return a;
}

// B fragment: 32x16 (KxN) taken from a row-major [K][N] source at p0.
// lanes 0-15: N=lane, elems j = K j ; lanes 16-31: N=lane-16, elems j = K 16+j
__device__ __forceinline__ v16h load_b_kn_lds(const _Float16* p0, int ld) {
  const int lane = threadIdx.x & 31;
  const _Float16* p = p0 + ((lane < 16) ? 0 : 16) * ld + (lane & 15);
  v16h b;
#pragma unroll
  for (int j = 0; j < 16; ++j) b[j] = p[j * ld];
  return b;
}

// B fragment built as the TRANSPOSE of a row-major [N][K] source:
// B[k][n] = src[n][k]  (contiguous 16-element K run per lane)
template <typename TP>
__device__ __forceinline__ v16h load_b_nk(const TP* p0, int ld) {
  const int lane = threadIdx.x & 31;
  const TP* p = p0 + (lane & 15) * ld + ((lane < 16) ? 0 : 16);
  v16h b;
#pragma unroll
  for (int j = 0; j < 16; ++j) b[j] = (_Float16)p[j];
  return b;
}

#define WMMA_F16(a, b, c) \
  __builtin_amdgcn_wmma_f32_16x16x32_f16(false, (a), false, (b), (short)0, (c), false, false)

// ---------- prep: W1/W2 fp32 -> f16 in workspace ----------
__global__ void convert_w_f16(const float* __restrict__ W1,
                              const float* __restrict__ W2,
                              _Float16* __restrict__ out) {
  int i = blockIdx.x * 256 + threadIdx.x;
  if (i < E_ * E_)            out[i] = (_Float16)W1[i];
  else if (i < 2 * E_ * E_)   out[i] = (_Float16)W2[i - E_ * E_];
}

// ---------- fused attention + FFN, one workgroup per (b,t,d) ----------
__global__ __launch_bounds__(256) void fused_attn_ffn(
    const float* __restrict__ Qg, const float* __restrict__ Kg,
    const float* __restrict__ Vg, const float* __restrict__ Mg,
    const float* __restrict__ W1g, const float* __restrict__ b1g,
    const float* __restrict__ W2g, const float* __restrict__ b2g,
    const float* __restrict__ ln1w, const float* __restrict__ ln1b,
    const float* __restrict__ ln2w, const float* __restrict__ ln2b,
    const _Float16* __restrict__ W1h, const _Float16* __restrict__ W2h,
    int useF16W, float* __restrict__ Og) {
  // LDS carve-out: 3x (64x256 f16) + (64x64 f16) + (50x256 f32) = 157696 B
  __shared__ __align__(16) char smem[157696];
  _Float16* sQ  = (_Float16*)smem;                // Q tile (64x256, padded)
  _Float16* sK  = (_Float16*)(smem + 32768);      // K tile -> reused as X(f16)
  _Float16* sV  = (_Float16*)(smem + 65536);      // V tile -> reused as H(f16)
  _Float16* sS  = (_Float16*)(smem + 98304);      // 64x64 f16 softmax scores
  float*    sXf = (float*)(smem + 106496);        // 50x256 f32 (x / ffn+x)
  float*    sLg = sXf;                            // alias: 64x64 f32 logits

  const int tid  = threadIdx.x;
  const int wave = tid >> 5;
  const int lane = tid & 31;
  const int bid  = blockIdx.x;
  const int d = bid % D_;
  const int t = (bid / D_) % T_;
  const int b = bid / (D_ * T_);

  const float* Qbase = Qg + (size_t)(b * T_ + t) * U_ * E_;
  const float* Kbase = Kg + (size_t)(b * D_ + d) * P_ * E_;
  const float* Vbase = Vg + (size_t)(b * D_ + d) * P_ * E_;

  // ---- Stage 0: global f32 -> LDS f16, zero-pad rows to 64 ----
  for (int s = tid; s < 64 * 64; s += 256) {       // 64 rows x 64 float4
    int row = s >> 6;
    int c4  = (s & 63) << 2;
    float4 q = {0.f, 0.f, 0.f, 0.f}, k = q, v = q;
    if (row < U_) q = *(const float4*)(Qbase + row * E_ + c4);
    if (row < P_) {
      k = *(const float4*)(Kbase + row * E_ + c4);
      v = *(const float4*)(Vbase + row * E_ + c4);
    }
    *(h4*)(sQ + row * E_ + c4) = (h4){(_Float16)q.x, (_Float16)q.y, (_Float16)q.z, (_Float16)q.w};
    *(h4*)(sK + row * E_ + c4) = (h4){(_Float16)k.x, (_Float16)k.y, (_Float16)k.z, (_Float16)k.w};
    *(h4*)(sV + row * E_ + c4) = (h4){(_Float16)v.x, (_Float16)v.y, (_Float16)v.z, (_Float16)v.w};
  }
  __syncthreads();

  // ---- Stage 1: logits = Q K^T / scale   (64x64, 4x4 tiles, 2 per wave) ----
  const float inv_scale = 1.0f / (16.0f + 1e-8f);  // sqrt(256)+eps
#pragma unroll
  for (int i = 0; i < 2; ++i) {
    int idx = wave * 2 + i;
    int mi = idx >> 2, ni = idx & 3;
    v8f acc = {};
#pragma unroll
    for (int k0 = 0; k0 < 8; ++k0) {
      v16h a  = load_a_lds(sQ + mi * 16 * E_ + k0 * 32, E_);
      v16h bb = load_b_nk(sK + ni * 16 * E_ + k0 * 32, E_);   // B = K^T
      acc = WMMA_F16(a, bb, acc);
    }
    int col   = ni * 16 + (lane & 15);
    int rbase = mi * 16 + ((lane < 16) ? 0 : 8);
#pragma unroll
    for (int v2 = 0; v2 < 8; ++v2)
      sLg[(rbase + v2) * 64 + col] = acc[v2] * inv_scale;
  }
  __syncthreads();

  // ---- Softmax over P (one thread per row) ----
  if (tid < 64) {
    int u = tid;
    if (u < U_) {
      const float* mrow = Mg + ((size_t)((b * T_ + t) * D_ + d) * U_ + u) * P_;
      float mx = -3.4e38f;
      for (int p = 0; p < P_; ++p) mx = fmaxf(mx, sLg[u * 64 + p] + mrow[p]);
      float sum = 0.f;
      for (int p = 0; p < P_; ++p) sum += __expf(sLg[u * 64 + p] + mrow[p] - mx);
      float inv = 1.f / sum;
      for (int p = 0; p < P_; ++p)
        sS[u * 64 + p] = (_Float16)(__expf(sLg[u * 64 + p] + mrow[p] - mx) * inv);
      for (int p = P_; p < 64; ++p) sS[u * 64 + p] = (_Float16)0.f;
    } else {
      for (int p = 0; p < 64; ++p) sS[u * 64 + p] = (_Float16)0.f;
    }
  }
  __syncthreads();

  // ---- Stage 2: x = Q + S V   (4x16 tiles, 8 per wave) ----
#pragma unroll
  for (int i = 0; i < 8; ++i) {
    int idx = wave * 8 + i;
    int mi = idx >> 4, ni = idx & 15;
    v8f acc = {};
#pragma unroll
    for (int k0 = 0; k0 < 2; ++k0) {
      v16h a  = load_a_lds(sS + mi * 16 * 64 + k0 * 32, 64);
      v16h bb = load_b_kn_lds(sV + (k0 * 32) * E_ + ni * 16, E_);
      acc = WMMA_F16(a, bb, acc);
    }
    int col   = ni * 16 + (lane & 15);
    int rbase = mi * 16 + ((lane < 16) ? 0 : 8);
#pragma unroll
    for (int v2 = 0; v2 < 8; ++v2) {
      int row = rbase + v2;
      if (row < U_) sXf[row * E_ + col] = acc[v2] + Qbase[row * E_ + col];
    }
  }
  __syncthreads();

  // ---- LN1: one wave per row; f32 residual + f16 GEMM operand (into sK) ----
  for (int u = wave; u < U_; u += 8) {
    float x[8], ssum = 0.f;
#pragma unroll
    for (int j = 0; j < 8; ++j) { x[j] = sXf[u * E_ + lane + j * 32]; ssum += x[j]; }
    float mean = wave_sum(ssum) * (1.f / E_);
    float vsum = 0.f;
#pragma unroll
    for (int j = 0; j < 8; ++j) { float dx = x[j] - mean; vsum += dx * dx; }
    float rstd = rsqrtf(wave_sum(vsum) * (1.f / E_) + 1e-5f);
#pragma unroll
    for (int j = 0; j < 8; ++j) {
      int c = lane + j * 32;
      float xn = (x[j] - mean) * rstd * ln1w[c] + ln1b[c];
      sXf[u * E_ + c] = xn;
      sK[u * E_ + c]  = (_Float16)xn;
    }
  }
  for (int s = tid; s < (64 - U_) * E_; s += 256) sK[U_ * E_ + s] = (_Float16)0.f;
  __syncthreads();

  // ---- Stage 3: H = relu(X W1^T + b1)  (into sV) ----
#pragma unroll
  for (int i = 0; i < 8; ++i) {
    int idx = wave * 8 + i;
    int mi = idx >> 4, ni = idx & 15;
    v8f acc = {};
    if (useF16W) {
#pragma unroll
      for (int k0 = 0; k0 < 8; ++k0) {
        v16h a  = load_a_lds(sK + mi * 16 * E_ + k0 * 32, E_);
        v16h bb = load_b_nk(W1h + ni * 16 * E_ + k0 * 32, E_);  // B = W1^T
        acc = WMMA_F16(a, bb, acc);
      }
    } else {
#pragma unroll
      for (int k0 = 0; k0 < 8; ++k0) {
        v16h a  = load_a_lds(sK + mi * 16 * E_ + k0 * 32, E_);
        v16h bb = load_b_nk(W1g + ni * 16 * E_ + k0 * 32, E_);
        acc = WMMA_F16(a, bb, acc);
      }
    }
    int col   = ni * 16 + (lane & 15);
    int rbase = mi * 16 + ((lane < 16) ? 0 : 8);
    float bias = b1g[col];
#pragma unroll
    for (int v2 = 0; v2 < 8; ++v2)
      sV[(rbase + v2) * E_ + col] = (_Float16)fmaxf(acc[v2] + bias, 0.f);
  }
  __syncthreads();

  // ---- Stage 4: F = H W2^T + b2; residual accumulate into sXf ----
#pragma unroll
  for (int i = 0; i < 8; ++i) {
    int idx = wave * 8 + i;
    int mi = idx >> 4, ni = idx & 15;
    v8f acc = {};
    if (useF16W) {
#pragma unroll
      for (int k0 = 0; k0 < 8; ++k0) {
        v16h a  = load_a_lds(sV + mi * 16 * E_ + k0 * 32, E_);
        v16h bb = load_b_nk(W2h + ni * 16 * E_ + k0 * 32, E_);
        acc = WMMA_F16(a, bb, acc);
      }
    } else {
#pragma unroll
      for (int k0 = 0; k0 < 8; ++k0) {
        v16h a  = load_a_lds(sV + mi * 16 * E_ + k0 * 32, E_);
        v16h bb = load_b_nk(W2g + ni * 16 * E_ + k0 * 32, E_);
        acc = WMMA_F16(a, bb, acc);
      }
    }
    int col   = ni * 16 + (lane & 15);
    int rbase = mi * 16 + ((lane < 16) ? 0 : 8);
    float bias = b2g[col];
#pragma unroll
    for (int v2 = 0; v2 < 8; ++v2) {
      int row = rbase + v2;
      if (row < U_) sXf[row * E_ + col] += acc[v2] + bias;
    }
  }
  __syncthreads();

  // ---- LN2 + coalesced store ----
  float* obase = Og + (size_t)bid * U_ * E_;   // bid == ((b*T+t)*D+d)
  for (int u = wave; u < U_; u += 8) {
    float x[8], ssum = 0.f;
#pragma unroll
    for (int j = 0; j < 8; ++j) { x[j] = sXf[u * E_ + lane + j * 32]; ssum += x[j]; }
    float mean = wave_sum(ssum) * (1.f / E_);
    float vsum = 0.f;
#pragma unroll
    for (int j = 0; j < 8; ++j) { float dx = x[j] - mean; vsum += dx * dx; }
    float rstd = rsqrtf(wave_sum(vsum) * (1.f / E_) + 1e-5f);
#pragma unroll
    for (int j = 0; j < 8; ++j) {
      int c = lane + j * 32;
      obase[u * E_ + c] = (x[j] - mean) * rstd * ln2w[c] + ln2b[c];
    }
  }
}

extern "C" void kernel_launch(void* const* d_in, const int* in_sizes, int n_in,
                              void* d_out, int out_size, void* d_ws, size_t ws_size,
                              hipStream_t stream) {
  const float* Qg   = (const float*)d_in[0];
  const float* Kg   = (const float*)d_in[1];
  const float* Vg   = (const float*)d_in[2];
  const float* Mg   = (const float*)d_in[3];
  const float* W1g  = (const float*)d_in[4];
  const float* b1g  = (const float*)d_in[5];
  const float* W2g  = (const float*)d_in[6];
  const float* b2g  = (const float*)d_in[7];
  const float* ln1w = (const float*)d_in[8];
  const float* ln1b = (const float*)d_in[9];
  const float* ln2w = (const float*)d_in[10];
  const float* ln2b = (const float*)d_in[11];
  float* Og = (float*)d_out;

  _Float16* Wh = (_Float16*)d_ws;
  int useF16W = (ws_size >= (size_t)2 * E_ * E_ * sizeof(_Float16)) ? 1 : 0;
  if (useF16W)
    convert_w_f16<<<(2 * E_ * E_) / 256, 256, 0, stream>>>(W1g, W2g, Wh);

  fused_attn_ffn<<<B_ * T_ * D_, 256, 0, stream>>>(
      Qg, Kg, Vg, Mg, W1g, b1g, W2g, b2g, ln1w, ln1b, ln2w, ln2b,
      Wh, Wh + E_ * E_, useF16W, Og);
}
